// MultiHeadAttention_1838246003069
// MI455X (gfx1250) — compile-verified, matrix operand order checked by reading, not run on hardware
//
#include <hip/hip_runtime.h>
#include <hip/hip_bf16.h>

// ---------------------------------------------------------------------------
// MultiHeadAttention forward on gfx1250 (MI455X), WMMA f16 path.
// B=2, S=2048, D=1024, H=16, HD=64.  Outputs: x [B,S,D] fp32 then attn
// [B,H,S,S] fp32, concatenated in d_out.
//
// Roofline: ~69 GFLOP total, but the mandatory 537 MB fp32 attn write is the
// floor (~24 us @ 23.3 TB/s).  All GEMMs use v_wmma_f32_16x16x32_f16 (f32
// accumulate); attn writes are non-temporal so the resident K/V/Q (24 MB)
// stay in the 192 MB L2 across the 128 q-tile blocks that re-read them.
// ---------------------------------------------------------------------------

typedef __attribute__((ext_vector_type(16))) _Float16 v16h;
typedef __attribute__((ext_vector_type(8)))  _Float16 v8h;
typedef __attribute__((ext_vector_type(8)))  float    v8f;
typedef __attribute__((ext_vector_type(4)))  float    v4f;

#define NB   2
#define NS   2048
#define ND   1024
#define NH   16
#define NHD  64
#define SP   2056   // LDS row stride (floats) for the 16 x 2048 score tile

// 16x32 f16 operand fragment from a row-major tile (row stride ld halves).
// Lane l: row = l&15, half h = l>>4.  Elements 0..7 <- K = h*8+{0..7},
// elements 8..15 <- K = 16+h*8+{0..7}  (CDNA5 ISA 7.12.2, 16-bit A 16x32).
// The B operand uses the mirrored (column-indexed) layout, so it loads the
// same way from B^T stored row-major.
__device__ __forceinline__ v16h load_frag16(const _Float16* __restrict__ base, int ld) {
  const int lane = threadIdx.x & 31;
  const int r  = lane & 15;
  const int hh = lane >> 4;
  const _Float16* p = base + r * ld + hh * 8;
  v8h lo = *(const v8h*)(p);
  v8h hi = *(const v8h*)(p + 16);
  v16h f;
#pragma unroll
  for (int i = 0; i < 8; ++i) { f[i] = lo[i]; f[i + 8] = hi[i]; }
  return f;
}

__device__ __forceinline__ v8f wmma32(v16h a, v16h b, v8f c) {
  return __builtin_amdgcn_wmma_f32_16x16x32_f16(false, a, false, b,
                                                (short)0, c, false, false);
}

// ---------------------------------------------------------------------------
// fp32 -> f16 convert; inputs are read exactly once -> non-temporal loads.
__global__ void cvt_f32_f16(const float* __restrict__ s,
                            _Float16* __restrict__ d, int n) {
  int i = blockIdx.x * blockDim.x + threadIdx.x;
  int stride = gridDim.x * blockDim.x;
  for (; i < n; i += stride) d[i] = (_Float16)__builtin_nontemporal_load(s + i);
}

// ---------------------------------------------------------------------------
// QKV projection: y = x @ W^T + b.  blockIdx.z: 0=Q, 1=K, 2=V.
// Q,K -> [B,H,S,HD] f16 (Q pre-scaled by 1/8); V -> [B,H,HD,S] f16.
// 8 waves/block; each wave computes a 16x32 strip (two N-tiles share one A
// fragment -> half the A traffic, 2 WMMAs per loaded A).  K = 1024.
__global__ __launch_bounds__(256) void qkv_gemm(
    const _Float16* __restrict__ xq, const _Float16* __restrict__ xk,
    const _Float16* __restrict__ xv,
    const _Float16* __restrict__ Wq, const _Float16* __restrict__ Wk,
    const _Float16* __restrict__ Wv,
    const float* __restrict__ bq, const float* __restrict__ bk,
    const float* __restrict__ bv,
    _Float16* __restrict__ Qh, _Float16* __restrict__ Kh,
    _Float16* __restrict__ VT) {
  const int z    = blockIdx.z;
  const int wave = threadIdx.x >> 5;
  const int lane = threadIdx.x & 31;
  const int t    = blockIdx.x * 8 + wave;       // 1024 blocks * 8 = 8192 strips
  const int mt   = t >> 5;                      // 256 M-tiles over B*S
  const int np   = t & 31;                      // 32 N-pairs over D

  const _Float16* X = (z == 0) ? xq : (z == 1) ? xk : xv;
  const _Float16* W = (z == 0) ? Wq : (z == 1) ? Wk : Wv;
  const float*  bias = (z == 0) ? bq : (z == 1) ? bk : bv;

  const _Float16* Ab  = X + (size_t)mt * 16 * ND;
  const _Float16* Bb0 = W + (size_t)np * 32 * ND;   // W row-major == B^T
  const _Float16* Bb1 = Bb0 + (size_t)16 * ND;

  v8f acc0 = {}, acc1 = {};
  for (int k0 = 0; k0 < ND; k0 += 32) {
    v16h a  = load_frag16(Ab + k0, ND);
    v16h b0 = load_frag16(Bb0 + k0, ND);
    v16h b1 = load_frag16(Bb1 + k0, ND);
    acc0 = wmma32(a, b0, acc0);
    acc1 = wmma32(a, b1, acc1);
  }

  const int  N  = lane & 15;
  const int  mo = (lane >> 4) * 8;
  const float sc = (z == 0) ? 0.125f : 1.0f;     // fold 1/sqrt(HD) into Q
#pragma unroll
  for (int half = 0; half < 2; ++half) {
    const v8f& acc = half ? acc1 : acc0;
    const int  e   = np * 32 + half * 16 + N;
    const float bb = bias[e];
    const int head = e >> 6, hd = e & 63;
#pragma unroll
    for (int r = 0; r < 8; ++r) {
      const int m  = mt * 16 + mo + r;           // global row over B*S
      const int b_ = m >> 11, s = m & 2047;
      const _Float16 val = (_Float16)((acc[r] + bb) * sc);
      if (z == 2) {
        VT[((size_t)(b_ * NH + head) * NHD + hd) * NS + s] = val;
      } else {
        _Float16* dst = (z == 0) ? Qh : Kh;
        dst[((size_t)(b_ * NH + head) * NS + s) * NHD + hd] = val;
      }
    }
  }
}

// ---------------------------------------------------------------------------
// Fused attention: one block per (b, h, 16-row q-tile).  8 waves.
// Phase A: scores = Q K^T (Q pre-scaled) -> LDS (16 x 2048 fp32) + mask.
// Phase B: row softmax (wave32 shuffle reductions), attn -> global fp32 (NT).
// Phase C: P @ V via WMMA (P cvt fp32->f16 from LDS), context -> f16.
__global__ __launch_bounds__(256) void attn_fused(
    const _Float16* __restrict__ Qh, const _Float16* __restrict__ Kh,
    const _Float16* __restrict__ VT, const int* __restrict__ mask,
    float* __restrict__ attnOut, _Float16* __restrict__ ctx16) {
  extern __shared__ float smem[];
  float* scLds = smem;                  // 16 * SP floats
  float* pbuf  = smem + 16 * SP;        // 8 * 256 floats (PV partials)

  const int wave = threadIdx.x >> 5;
  const int lane = threadIdx.x & 31;
  const int bid  = blockIdx.x;          // 4096 = B*H*(S/16)
  const int qt   = bid & 127;
  const int h    = (bid >> 7) & 15;
  const int b    = bid >> 11;
  const size_t headOff = (size_t)(b * NH + h);

  const int N  = lane & 15;
  const int mo = (lane >> 4) * 8;

  // ---- Phase A: score tile -------------------------------------------------
  const _Float16* Qb = Qh + (headOff * NS + (size_t)qt * 16) * NHD;
  const v16h q0 = load_frag16(Qb, NHD);        // K = 0..31 of HD
  const v16h q1 = load_frag16(Qb + 32, NHD);   // K = 32..63
  const _Float16* Kbase = Kh + headOff * NS * NHD;

  for (int nt = wave; nt < NS / 16; nt += 8) {
    const _Float16* Kb = Kbase + (size_t)nt * 16 * NHD;
    v16h k0f = load_frag16(Kb, NHD);
    v16h k1f = load_frag16(Kb + 32, NHD);
    v8f acc = {};
    acc = wmma32(q0, k0f, acc);
    acc = wmma32(q1, k1f, acc);
    const int mv = mask[b * NS + nt * 16 + N];
#pragma unroll
    for (int r = 0; r < 8; ++r) {
      const float v = (mv == 0) ? -1e9f : acc[r];
      scLds[(mo + r) * SP + nt * 16 + N] = v;
    }
  }
  __syncthreads();

  // ---- Phase B: softmax (2 rows per wave) ---------------------------------
  for (int rr = 0; rr < 2; ++rr) {
    const int row = wave + rr * 8;
    float* rowp = scLds + row * SP;
    float mx = -3.4e38f;
    for (int j = lane; j < NS; j += 32) mx = fmaxf(mx, rowp[j]);
#pragma unroll
    for (int o = 16; o >= 1; o >>= 1) mx = fmaxf(mx, __shfl_xor(mx, o, 32));
    float sum = 0.f;
    for (int j = lane; j < NS; j += 32) {
      const float e = __expf(rowp[j] - mx);
      rowp[j] = e;
      sum += e;
    }
#pragma unroll
    for (int o = 16; o >= 1; o >>= 1) sum += __shfl_xor(sum, o, 32);
    const float inv = 1.0f / sum;
    float* aout = attnOut + (headOff * NS + (size_t)qt * 16 + row) * NS;
    for (int j = lane; j < NS; j += 32) {
      const float p = rowp[j] * inv;
      rowp[j] = p;
      __builtin_nontemporal_store(p, aout + j);  // 537 MB stream: keep out of L2
    }
  }
  __syncthreads();

  // ---- Phase C: context = P @ V  (4 hd-tiles x 2 K-halves over 8 waves) ---
  const int nt2 = wave & 3;             // 16-wide tile of HD
  const int kh2 = wave >> 2;            // K half: 0 -> [0,1024), 1 -> [1024,2048)
  const _Float16* Vb = VT + (headOff * NHD + (size_t)nt2 * 16) * NS;
  const int mrow = lane & 15;
  const int hh   = lane >> 4;
  v8f acc = {};
  for (int k0 = kh2 * 1024; k0 < kh2 * 1024 + 1024; k0 += 32) {
    const float* p = scLds + mrow * SP + k0 + hh * 8;
    v4f a0 = *(const v4f*)(p);
    v4f a1 = *(const v4f*)(p + 4);
    v4f a2 = *(const v4f*)(p + 16);
    v4f a3 = *(const v4f*)(p + 20);
    v16h af;
#pragma unroll
    for (int i = 0; i < 4; ++i) {
      af[i]      = (_Float16)a0[i];
      af[4 + i]  = (_Float16)a1[i];
      af[8 + i]  = (_Float16)a2[i];
      af[12 + i] = (_Float16)a3[i];
    }
    v16h bf = load_frag16(Vb + k0, NS);
    acc = wmma32(af, bf, acc);
  }
#pragma unroll
  for (int r = 0; r < 8; ++r) pbuf[wave * 256 + r * 32 + lane] = acc[r];
  __syncthreads();
  if (wave < 4) {
#pragma unroll
    for (int r = 0; r < 8; ++r) {
      const float v = pbuf[wave * 256 + r * 32 + lane] +
                      pbuf[(wave + 4) * 256 + r * 32 + lane];
      const int m = qt * 16 + mo + r;
      ctx16[((size_t)(b * NS + m)) * ND + h * NHD + nt2 * 16 + N] = (_Float16)v;
    }
  }
}

// ---------------------------------------------------------------------------
// Output projection: x = ctx @ Wo^T + bo  (fp32 out).  Same 16x32-per-wave
// strip mining as qkv_gemm.
__global__ __launch_bounds__(256) void out_gemm(
    const _Float16* __restrict__ ctx16, const _Float16* __restrict__ Wo,
    const float* __restrict__ bo, float* __restrict__ xout) {
  const int wave = threadIdx.x >> 5;
  const int lane = threadIdx.x & 31;
  const int t    = blockIdx.x * 8 + wave;
  const int mt   = t >> 5;
  const int np   = t & 31;

  const _Float16* Ab  = ctx16 + (size_t)mt * 16 * ND;
  const _Float16* Bb0 = Wo + (size_t)np * 32 * ND;
  const _Float16* Bb1 = Bb0 + (size_t)16 * ND;

  v8f acc0 = {}, acc1 = {};
  for (int k0 = 0; k0 < ND; k0 += 32) {
    v16h a  = load_frag16(Ab + k0, ND);
    v16h b0 = load_frag16(Bb0 + k0, ND);
    v16h b1 = load_frag16(Bb1 + k0, ND);
    acc0 = wmma32(a, b0, acc0);
    acc1 = wmma32(a, b1, acc1);
  }
  const int N  = lane & 15;
  const int mo = (lane >> 4) * 8;
#pragma unroll
  for (int half = 0; half < 2; ++half) {
    const v8f& acc = half ? acc1 : acc0;
    const int  e   = np * 32 + half * 16 + N;
    const float bb = bo[e];
#pragma unroll
    for (int r = 0; r < 8; ++r) {
      const int m = mt * 16 + mo + r;
      xout[(size_t)m * ND + e] = acc[r] + bb;
    }
  }
}

// ---------------------------------------------------------------------------
extern "C" void kernel_launch(void* const* d_in, const int* in_sizes, int n_in,
                              void* d_out, int out_size, void* d_ws,
                              size_t ws_size, hipStream_t stream) {
  const float* query = (const float*)d_in[0];
  const float* key   = (const float*)d_in[1];
  const float* value = (const float*)d_in[2];
  const int*   mask  = (const int*)d_in[3];
  const float* Wq = (const float*)d_in[4];
  const float* bq = (const float*)d_in[5];
  const float* Wk = (const float*)d_in[6];
  const float* bk = (const float*)d_in[7];
  const float* Wv = (const float*)d_in[8];
  const float* bv = (const float*)d_in[9];
  const float* Wo = (const float*)d_in[10];
  const float* bo = (const float*)d_in[11];

  const int NX = NB * NS * ND;   // 4,194,304
  const int NW = ND * ND;        // 1,048,576

  // Workspace layout (f16 elements), total 33,554,432 halves = 64 MB.
  _Float16* ws   = (_Float16*)d_ws;
  _Float16* xq16 = ws;
  _Float16* xk16 = xq16 + NX;
  _Float16* xv16 = xk16 + NX;
  _Float16* Wq16 = xv16 + NX;
  _Float16* Wk16 = Wq16 + NW;
  _Float16* Wv16 = Wk16 + NW;
  _Float16* Wo16 = Wv16 + NW;
  _Float16* Qh   = Wo16 + NW;    // [B,H,S,HD]
  _Float16* Kh   = Qh + NX;      // [B,H,S,HD]
  _Float16* VT   = Kh + NX;      // [B,H,HD,S]
  _Float16* ctx  = VT + NX;      // [B*S, D]

  float* xout    = (float*)d_out;
  float* attnOut = xout + NX;    // attn [B,H,S,S] follows x

  cvt_f32_f16<<<4096, 256, 0, stream>>>(query, xq16, NX);
  cvt_f32_f16<<<4096, 256, 0, stream>>>(key,   xk16, NX);
  cvt_f32_f16<<<4096, 256, 0, stream>>>(value, xv16, NX);
  cvt_f32_f16<<<2048, 256, 0, stream>>>(Wq, Wq16, NW);
  cvt_f32_f16<<<2048, 256, 0, stream>>>(Wk, Wk16, NW);
  cvt_f32_f16<<<2048, 256, 0, stream>>>(Wv, Wv16, NW);
  cvt_f32_f16<<<2048, 256, 0, stream>>>(Wo, Wo16, NW);

  qkv_gemm<<<dim3(1024, 1, 3), 256, 0, stream>>>(
      xq16, xk16, xv16, Wq16, Wk16, Wv16, bq, bk, bv, Qh, Kh, VT);

  const size_t smem = (size_t)(16 * SP + 8 * 256) * sizeof(float); // ~136.5 KB
  (void)hipFuncSetAttribute((const void*)attn_fused,
                            hipFuncAttributeMaxDynamicSharedMemorySize,
                            (int)smem);
  attn_fused<<<NB * NH * (NS / 16), 256, smem, stream>>>(
      Qh, Kh, VT, mask, attnOut, ctx);

  out_gemm<<<1024, 256, 0, stream>>>(ctx, Wo16, bo, xout);
}
